// Local_Inference_Modeling_18554258719076
// MI455X (gfx1250) — compile-verified, matrix operand order checked by reading, not run on hardware
//
#include <hip/hip_runtime.h>

// ---------------------------------------------------------------------------
// ESIM-style cross attention for MI455X (gfx1250), wave32 + WMMA bf16 + TDM.
//   B=32, La=Lb=512, D=768.
//   cvt->bf16 | scores GEMM (TDM-staged) | row softmax | col softmax (transposed)
//   | attend GEMM (TDM-staged P, transposed-staged V) + concat epilogue x2.
// All intermediates live in d_ws (~112 MB -> resident in 192 MB L2).
// ---------------------------------------------------------------------------

typedef __attribute__((ext_vector_type(16))) __bf16       v16bf;
typedef __attribute__((ext_vector_type(8)))  __bf16       v8bf;
typedef __attribute__((ext_vector_type(4)))  __bf16       v4bf;
typedef __attribute__((ext_vector_type(2)))  __bf16       v2bf;
typedef __attribute__((ext_vector_type(8)))  float        v8f;
typedef __attribute__((ext_vector_type(4)))  unsigned int u32x4;
typedef __attribute__((ext_vector_type(8)))  unsigned int u32x8;

#define NB   32
#define LSEQ 512
#define DDIM 768
#define LDSTRIDE 40                 // 32 k-elems + 8 pad (80B rows, 16B aligned)
#define TILEBYTES (128 * LDSTRIDE * 2)   // one LDS tile buffer (10240 B)

// ---------------------------------------------------------------------------
// TDM: DMA one 128x32 bf16 tile (row-major, row stride `row_stride` elems)
// from global into LDS at byte offset `lds_addr`, inserting 16B of padding
// after every 64B row so the LDS image has LDSTRIDE=40 element rows.
// D# group0/group1 per CDNA5 ISA 8.3/8.4; groups 2/3 omitted (2D tensor).
// Issued by one wave; completion tracked with TENSORcnt.
// ---------------------------------------------------------------------------
__device__ inline void tdm_load_tile(unsigned int lds_addr, const void* gptr,
                                     unsigned int row_stride_elems) {
  unsigned long long ga = (unsigned long long)(uintptr_t)gptr;
  u32x4 g0;
  g0[0] = 1u;                                   // count=1, user descriptor
  g0[1] = lds_addr;                             // LDS byte address
  g0[2] = (unsigned int)ga;                     // global_addr[31:0]
  g0[3] = (unsigned int)(ga >> 32) | (2u << 30);// global_addr[56:32] | type=2
  u32x8 g1;
  g1[0] = (1u << 16)                            // data_size = 2 bytes
        | (1u << 20)                            // pad_enable
        | (3u << 22)                            // pad_interval: 16 DWORDs (64B)
        | (3u << 25);                           // pad_amount:   4 DWORDs (16B)
  g1[1] = (0xFFFFFu & 0xFFFFu) << 16;           // tensor_dim0[15:0] (huge)
  g1[2] = (0xFFFFFu >> 16) | ((0xFFFFFu & 0xFFFFu) << 16); // dim0 hi | dim1 lo
  g1[3] = (0xFFFFFu >> 16) | (32u << 16);       // dim1 hi | tile_dim0 = 32
  g1[4] = 128u;                                 // tile_dim1 = 128, tile_dim2 = 0
  g1[5] = row_stride_elems;                     // tensor_dim0_stride[31:0]
  g1[6] = 0u;                                   // stride0 hi | stride1 lo
  g1[7] = 0u;                                   // stride1 hi
  asm volatile("tensor_load_to_lds %0, %1" :: "s"(g0), "s"(g1) : "memory");
}

// Build a 16x32 bf16 WMMA A/B fragment per the CDNA5 16-bit layout:
// lane<16 : K {0..7, 16..23}; lane>=16 : K {8..15, 24..31}.
__device__ inline v16bf load_frag(const __bf16* p) {
  v8bf lo = *(const v8bf*)(p);
  v8bf hi = *(const v8bf*)(p + 16);
  return __builtin_shufflevector(lo, hi, 0, 1, 2, 3, 4, 5, 6, 7,
                                         8, 9, 10, 11, 12, 13, 14, 15);
}

// ---------------------------------------------------------------------------
// K0: f32 -> bf16 conversion (4 elems/thread)
// ---------------------------------------------------------------------------
__global__ __launch_bounds__(256)
void cvt_bf16_kernel(const float* __restrict__ in, __bf16* __restrict__ out, int n4) {
  int i = blockIdx.x * blockDim.x + threadIdx.x;
  if (i < n4) {
    float4 f = *(const float4*)(in + (size_t)i * 4);
    v4bf o;
    o[0] = (__bf16)f.x; o[1] = (__bf16)f.y; o[2] = (__bf16)f.z; o[3] = (__bf16)f.w;
    *(v4bf*)(out + (size_t)i * 4) = o;
  }
}

// ---------------------------------------------------------------------------
// K1: scores  e[b,a,c] = sum_d a16[b,a,d] * b16[b,c,d]
// grid (Lb/128, La/128, B); 8 waves; TDM double-buffered A/B tiles.
// ---------------------------------------------------------------------------
__global__ __launch_bounds__(256)
void scores_kernel(const __bf16* __restrict__ A16, const __bf16* __restrict__ B16,
                   float* __restrict__ E) {
  __shared__ __bf16 As[2][128 * LDSTRIDE];
  __shared__ __bf16 Bs[2][128 * LDSTRIDE];
  const int b = blockIdx.z;
  const int rowbase = blockIdx.y * 128;   // a index
  const int colbase = blockIdx.x * 128;   // c index
  const int tid  = threadIdx.x;
  const int lane = tid & 31;
  const int wave = tid >> 5;
  const int wrow = (wave & 3) * 32;
  const int wcol = (wave >> 2) * 64;
  const int lr   = lane & 15;
  const int half = lane >> 4;
  const bool issuer = (wave == 0);

  const __bf16* Ag = A16 + (size_t)b * LSEQ * DDIM + (size_t)rowbase * DDIM;
  const __bf16* Bg = B16 + (size_t)b * LSEQ * DDIM + (size_t)colbase * DDIM;
  const unsigned int ldsA = (unsigned int)(uintptr_t)&As[0][0];
  const unsigned int ldsB = (unsigned int)(uintptr_t)&Bs[0][0];

  v8f acc[2][4] = {};

  if (issuer) {                               // prologue: fill buffer 0
    tdm_load_tile(ldsA, Ag, DDIM);
    tdm_load_tile(ldsB, Bg, DDIM);
  }

  int buf = 0;
  for (int k0 = 0; k0 < DDIM; k0 += 32) {
    const bool more = (k0 + 32 < DDIM);
    if (issuer) {
      if (more) {                             // overlap next tile's DMA
        tdm_load_tile(ldsA + (buf ^ 1) * TILEBYTES, Ag + k0 + 32, DDIM);
        tdm_load_tile(ldsB + (buf ^ 1) * TILEBYTES, Bg + k0 + 32, DDIM);
        __builtin_amdgcn_s_wait_tensorcnt(2); // current buffer's 2 loads done
      } else {
        __builtin_amdgcn_s_wait_tensorcnt(0);
      }
    }
    __syncthreads();

    v16bf afr[2], bfr[4];
#pragma unroll
    for (int mi = 0; mi < 2; ++mi)
      afr[mi] = load_frag(&As[buf][(wrow + mi * 16 + lr) * LDSTRIDE + half * 8]);
#pragma unroll
    for (int ni = 0; ni < 4; ++ni)
      bfr[ni] = load_frag(&Bs[buf][(wcol + ni * 16 + lr) * LDSTRIDE + half * 8]);
#pragma unroll
    for (int mi = 0; mi < 2; ++mi)
#pragma unroll
      for (int ni = 0; ni < 4; ++ni)
        acc[mi][ni] = __builtin_amdgcn_wmma_f32_16x16x32_bf16(
            false, afr[mi], false, bfr[ni], (short)0, acc[mi][ni], false, false);
    __syncthreads();                          // buffer free for next TDM write
    buf ^= 1;
  }

  float* Eg = E + (size_t)b * LSEQ * LSEQ;
  const int cn = lr, mo = half * 8;
#pragma unroll
  for (int mi = 0; mi < 2; ++mi)
#pragma unroll
    for (int ni = 0; ni < 4; ++ni)
#pragma unroll
      for (int i = 0; i < 8; ++i) {
        size_t rg = rowbase + wrow + mi * 16 + mo + i;
        size_t cg = colbase + wcol + ni * 16 + cn;
        Eg[rg * LSEQ + cg] = acc[mi][ni][i];
      }
}

// ---------------------------------------------------------------------------
// K2: row softmax (axis=2). One wave per row. p_a bf16 [b, a, c].
// ---------------------------------------------------------------------------
__global__ __launch_bounds__(256)
void softmax_row_kernel(const float* __restrict__ E, __bf16* __restrict__ P) {
  const int b    = blockIdx.y;
  const int lane = threadIdx.x & 31;
  const int row  = blockIdx.x * 8 + (threadIdx.x >> 5);
  const float* er = E + ((size_t)b * LSEQ + row) * LSEQ;

  float v[16];
  float m = -3.4e38f;
#pragma unroll
  for (int i = 0; i < 16; ++i) { v[i] = er[lane + 32 * i]; m = fmaxf(m, v[i]); }
#pragma unroll
  for (int off = 16; off > 0; off >>= 1) m = fmaxf(m, __shfl_xor(m, off, 32));
  float s = 0.f;
#pragma unroll
  for (int i = 0; i < 16; ++i) { v[i] = __expf(v[i] - m); s += v[i]; }
#pragma unroll
  for (int off = 16; off > 0; off >>= 1) s += __shfl_xor(s, off, 32);
  float inv = 1.f / s;

  __bf16* pr = P + ((size_t)b * LSEQ + row) * LSEQ;
#pragma unroll
  for (int i = 0; i < 16; ++i) pr[lane + 32 * i] = (__bf16)(v[i] * inv);
}

// ---------------------------------------------------------------------------
// K2T: column softmax (axis=1), stored TRANSPOSED: p_bT[b, c, a] contiguous in a.
// grid (Lb/64, B); (4 row-strips) x (64 cols); online (m,s) + LDS combine.
// ---------------------------------------------------------------------------
__global__ __launch_bounds__(256)
void softmax_col_kernel(const float* __restrict__ E, __bf16* __restrict__ PT) {
  __shared__ float red[4][64];
  const int b     = blockIdx.y;
  const int cbase = blockIdx.x * 64;
  const int c     = threadIdx.x & 63;
  const int ri    = threadIdx.x >> 6;
  const float* ec = E + (size_t)b * LSEQ * LSEQ + cbase + c;

  float m = -3.4e38f, s = 0.f;
  for (int r = ri; r < LSEQ; r += 4) {
    float x  = ec[(size_t)r * LSEQ];
    float nm = fmaxf(m, x);
    s = s * __expf(m - nm) + __expf(x - nm);
    m = nm;
  }
  red[ri][c] = m;
  __syncthreads();
  float M = fmaxf(fmaxf(red[0][c], red[1][c]), fmaxf(red[2][c], red[3][c]));
  __syncthreads();
  red[ri][c] = s * __expf(m - M);
  __syncthreads();
  float S = red[0][c] + red[1][c] + red[2][c] + red[3][c];
  float inv = 1.f / S;

  __bf16* pc = PT + ((size_t)b * LSEQ + cbase + c) * LSEQ;
  for (int r = ri; r < LSEQ; r += 4)
    pc[r] = (__bf16)(__expf(ec[(size_t)r * LSEQ] - M) * inv);
}

// ---------------------------------------------------------------------------
// K3: attend + concat epilogue.
//   T[m,d] = sum_k P[m,k] * V[k,d];  Out[m, :] = [X | T | X-T | X*T]
// P: bf16 (MxK row-major) staged via TDM. V: bf16 (KxN, N contiguous) staged
// manually with an LDS transpose ([n][k]). Double-buffered, grid (6,4,B).
// ---------------------------------------------------------------------------
__global__ __launch_bounds__(256)
void attend_kernel(const __bf16* __restrict__ P, const __bf16* __restrict__ V,
                   const float* __restrict__ X, float* __restrict__ Out) {
  __shared__ __bf16 Ps[2][128 * LDSTRIDE];
  __shared__ __bf16 Vs[2][128 * LDSTRIDE];
  const int b = blockIdx.z;
  const int rowbase = blockIdx.y * 128;   // m tile
  const int colbase = blockIdx.x * 128;   // d tile
  const int tid  = threadIdx.x;
  const int lane = tid & 31;
  const int wave = tid >> 5;
  const int wrow = (wave & 3) * 32;
  const int wcol = (wave >> 2) * 64;
  const int lr   = lane & 15;
  const int half = lane >> 4;
  const bool issuer = (wave == 0);

  const __bf16* Pg = P + (size_t)b * LSEQ * LSEQ + (size_t)rowbase * LSEQ;
  const __bf16* Vg = V + (size_t)b * LSEQ * DDIM + colbase;
  const unsigned int ldsP = (unsigned int)(uintptr_t)&Ps[0][0];

  const int kk = (tid >> 4) * 2;          // 0,2,..,30
  const int n0 = (tid & 15) * 8;          // 0..120

  v8f acc[2][4] = {};

  // prologue: stage V tile 0 (transpose) + TDM P tile 0
  {
    v8bf r0 = *(const v8bf*)&Vg[(size_t)kk * DDIM + n0];
    v8bf r1 = *(const v8bf*)&Vg[(size_t)(kk + 1) * DDIM + n0];
#pragma unroll
    for (int t = 0; t < 8; ++t) {
      v2bf pr; pr[0] = r0[t]; pr[1] = r1[t];
      *(v2bf*)&Vs[0][(n0 + t) * LDSTRIDE + kk] = pr;
    }
  }
  if (issuer) tdm_load_tile(ldsP, Pg, LSEQ);

  int buf = 0;
  for (int k0 = 0; k0 < LSEQ; k0 += 32) {
    const bool more = (k0 + 32 < LSEQ);
    if (more) {
      // stage next V tile into the alternate buffer (LDS transpose)
      v8bf r0 = *(const v8bf*)&Vg[(size_t)(k0 + 32 + kk) * DDIM + n0];
      v8bf r1 = *(const v8bf*)&Vg[(size_t)(k0 + 32 + kk + 1) * DDIM + n0];
#pragma unroll
      for (int t = 0; t < 8; ++t) {
        v2bf pr; pr[0] = r0[t]; pr[1] = r1[t];
        *(v2bf*)&Vs[buf ^ 1][(n0 + t) * LDSTRIDE + kk] = pr;
      }
      if (k0 + 64 < LSEQ)                  // warm the V slab after that
        __builtin_prefetch(&Vg[(size_t)(k0 + 64 + kk) * DDIM + n0], 0, 3);
      if (issuer) {
        tdm_load_tile(ldsP + (buf ^ 1) * TILEBYTES, Pg + k0 + 32, LSEQ);
        __builtin_amdgcn_s_wait_tensorcnt(1);  // current P tile landed
      }
    } else if (issuer) {
      __builtin_amdgcn_s_wait_tensorcnt(0);
    }
    __syncthreads();

    v16bf afr[2], bfr[4];
#pragma unroll
    for (int mi = 0; mi < 2; ++mi)
      afr[mi] = load_frag(&Ps[buf][(wrow + mi * 16 + lr) * LDSTRIDE + half * 8]);
#pragma unroll
    for (int ni = 0; ni < 4; ++ni)
      bfr[ni] = load_frag(&Vs[buf][(wcol + ni * 16 + lr) * LDSTRIDE + half * 8]);
#pragma unroll
    for (int mi = 0; mi < 2; ++mi)
#pragma unroll
      for (int ni = 0; ni < 4; ++ni)
        acc[mi][ni] = __builtin_amdgcn_wmma_f32_16x16x32_bf16(
            false, afr[mi], false, bfr[ni], (short)0, acc[mi][ni], false, false);
    __syncthreads();
    buf ^= 1;
  }

  const float* Xb = X + (size_t)b * LSEQ * DDIM;
  float* Ob       = Out + (size_t)b * LSEQ * (4 * DDIM);
  const int cn = lr, mo = half * 8;
#pragma unroll
  for (int mi = 0; mi < 2; ++mi)
#pragma unroll
    for (int ni = 0; ni < 4; ++ni)
#pragma unroll
      for (int i = 0; i < 8; ++i) {
        size_t rg = rowbase + wrow + mi * 16 + mo + i;
        size_t dg = colbase + wcol + ni * 16 + cn;
        float t = acc[mi][ni][i];
        float x = Xb[rg * DDIM + dg];
        float* o = Ob + rg * (4 * DDIM);
        o[dg]            = x;
        o[DDIM + dg]     = t;
        o[2 * DDIM + dg] = x - t;
        o[3 * DDIM + dg] = x * t;
      }
}

// ---------------------------------------------------------------------------
extern "C" void kernel_launch(void* const* d_in, const int* in_sizes, int n_in,
                              void* d_out, int out_size, void* d_ws, size_t ws_size,
                              hipStream_t stream) {
  (void)in_sizes; (void)n_in; (void)out_size; (void)ws_size;
  const float* a_bar = (const float*)d_in[0];
  const float* b_bar = (const float*)d_in[1];
  float* out = (float*)d_out;

  const size_t nE = (size_t)NB * LSEQ * DDIM;      // 12,582,912 elems per input
  const size_t nS = (size_t)NB * LSEQ * LSEQ;      // 8,388,608 score elems

  // Workspace carve (~112 MB total; base is 256B-aligned, offsets keep 16B align)
  __bf16* a16 = (__bf16*)d_ws;                     // nE * 2 B
  __bf16* b16 = a16 + nE;                          // nE * 2 B
  float*  e   = (float*)(b16 + nE);                // nS * 4 B
  __bf16* pa  = (__bf16*)(e + nS);                 // nS * 2 B
  __bf16* pbT = pa + nS;                           // nS * 2 B

  const int n4 = (int)(nE / 4);
  cvt_bf16_kernel<<<(n4 + 255) / 256, 256, 0, stream>>>(a_bar, a16, n4);
  cvt_bf16_kernel<<<(n4 + 255) / 256, 256, 0, stream>>>(b_bar, b16, n4);

  scores_kernel<<<dim3(LSEQ / 128, LSEQ / 128, NB), 256, 0, stream>>>(a16, b16, e);

  softmax_row_kernel<<<dim3(LSEQ / 8, NB), 256, 0, stream>>>(e, pa);
  softmax_col_kernel<<<dim3(LSEQ / 64, NB), 256, 0, stream>>>(e, pbT);

  // m_a = [a | a~ | a-a~ | a*a~],  a~ = pa @ b
  attend_kernel<<<dim3(DDIM / 128, LSEQ / 128, NB), 256, 0, stream>>>(pa, b16, a_bar, out);
  // m_b = [b | b~ | b-b~ | b*b~],  b~ = pbT @ a
  attend_kernel<<<dim3(DDIM / 128, LSEQ / 128, NB), 256, 0, stream>>>(
      pbT, a16, b_bar, out + (size_t)NB * LSEQ * (4 * DDIM));
}
